// HungarianMatchingLoss_33887291965950
// MI455X (gfx1250) — matrix operand order repaired and checked by reading, not compiled, not run on hardware
//
#include <hip/hip_runtime.h>
#include <hip/hip_fp16.h>
#include <stdint.h>

// ---------------------------------------------------------------------------
// Hungarian matching loss, CDNA5 (gfx1250, wave32).
// Permutation scoring as f16 WMMA GEMM with reduced cost:
//   C~[b,q,j] = C[b,q,j] - C[b,5,j]  (q = 0..4)  ->  K = 30, pad to 32,
//   cost(b,p) = sum_q C~[b,q,perm_p[q]] + const(b)   (argmin-equivalent)
// One v_wmma_f32_16x16x32_f16 per 16x16 output tile.
// ---------------------------------------------------------------------------

#define NQ      6
#define NCL     10
#define BATCH   32768
#define NLAYER  7
#define NPERM   720
#define NTILE   45        // 720 / 16 N-tiles
#define NBT     2048      // 32768 / 16 batch tiles
#define EPSW    0.1f
#define W0      (10.0f / 9.5f)   // normalized class weight, classes 0..8
#define W9      (5.0f  / 9.5f)   // normalized class weight, class 9

typedef __attribute__((ext_vector_type(16))) _Float16 v16h;
typedef __attribute__((ext_vector_type(8)))  float    v8f;

// workspace layout (bytes)
#define OFF_BF    0         // B-fragments: 45 tiles * 512 f16 = 46080 B
#define OFF_PERM  46080     // 720 packed perms (u32, 6 nibbles) = 2880 B
#define OFF_TGT   48960     // argmax(targets): 32768*6 int16 = 393216 B

// p-th lexicographic permutation of {0..5}, packed 4 bits per position.
// Matches itertools.permutations ordering (factorial number system).
__device__ __forceinline__ uint32_t perm_pack(int p) {
    const int fact[6] = {120, 24, 6, 2, 1, 1};
    int avail = 0x3F;
    uint32_t packed = 0;
    #pragma unroll
    for (int q = 0; q < 6; ++q) {
        int d = p / fact[q]; p -= d * fact[q];
        int j = 0;
        #pragma unroll
        for (int bit = 0; bit < 6; ++bit) {
            if (avail & (1 << bit)) { if (d == 0) { j = bit; break; } --d; }
        }
        avail &= ~(1 << j);
        packed |= (uint32_t)j << (4 * q);
    }
    return packed;
}

// monotone float -> u32 key (ascending); gives total order incl. tie-break
__device__ __forceinline__ uint32_t fkey(float f) {
    uint32_t u = __float_as_uint(f);
    return (u & 0x80000000u) ? ~u : (u | 0x80000000u);
}

// ---------------------------------------------------------------------------
// hml_init: zero output accumulator, packed perm table, and the constant
// one-hot B fragments for V_WMMA_F32_16X16X32_F16.
// B fragment (32x16 f16): lane -> N = lane%16; lanes<16 hold K=0..15,
// lanes>=16 hold K=16..31 (consecutive K per lane). K = q*6+j, q<=4.
// ---------------------------------------------------------------------------
__global__ void hml_init(_Float16* __restrict__ bf,
                         uint32_t* __restrict__ permtab,
                         float* __restrict__ out) {
    int gid = blockIdx.x * blockDim.x + threadIdx.x;
    if (gid == 0) out[0] = 0.0f;
    if (gid < NPERM) permtab[gid] = perm_pack(gid);
    if (gid < NTILE * 512) {
        int i    = gid & 15;
        int lane = (gid >> 4) & 31;
        int t    = gid >> 9;
        int k    = ((lane < 16) ? 0 : 16) + i;   // K row of B, 0..31
        int n    = lane & 15;                     // N col of B
        float val = 0.0f;
        if (k < 30) {                             // q = 0..4 only
            int q = k / NQ, j = k - NQ * q;
            uint32_t pp = perm_pack(t * 16 + n);
            val = (((pp >> (4 * q)) & 0xF) == (uint32_t)j) ? 1.0f : 0.0f;
        }
        bf[gid] = (_Float16)val;
    }
}

// ---------------------------------------------------------------------------
// hml_tgt: tgt_ids = argmax(targets, -1), first max wins (jnp.argmax)
// ---------------------------------------------------------------------------
__global__ void hml_tgt(const float* __restrict__ targets,
                        short* __restrict__ tgt) {
    int gid = blockIdx.x * blockDim.x + threadIdx.x;
    if (gid >= BATCH * NQ) return;
    const float* row = targets + (size_t)gid * NCL;
    float bv = row[0]; int am = 0;
    #pragma unroll
    for (int j = 1; j < NCL; ++j) {
        float v = row[j];
        if (v > bv) { bv = v; am = j; }
    }
    tgt[gid] = (short)am;
}

// ---------------------------------------------------------------------------
// hml_main: 4 waves / block, each wave owns 16 batch rows across all layers.
// ---------------------------------------------------------------------------
__launch_bounds__(128)
__global__ void hml_main(const float* __restrict__ outputs,   // (7,32768,6,10)
                         const short* __restrict__ tgt,       // (32768,6)
                         const _Float16* __restrict__ bfrag,  // (45,32,16) f16
                         const uint32_t* __restrict__ permtab,
                         float* __restrict__ out) {
    __shared__ uint32_t s_perm[NPERM];
    __shared__ __align__(16) float s_logit[4][16][NQ * NCL];
    __shared__ float s_lse[4][16][NQ];
    __shared__ float s_swl[4][16][NQ];
    __shared__ __align__(16) float s_Craw[4][16][NQ * NQ];   // -p[b,q,tgt[b,j]]
    __shared__ short s_tgt[4][16][NQ];
    __shared__ float s_red[4];

    const int tid  = threadIdx.x;
    const int wave = tid >> 5;
    const int lane = tid & 31;
    const int tileB = blockIdx.x * 4 + wave;    // 0..2047
    const int b0 = tileB * 16;

    for (int i = tid; i < NPERM; i += 128) s_perm[i] = permtab[i];

    if (lane < 16) {
        #pragma unroll
        for (int j = 0; j < NQ; ++j)
            s_tgt[wave][lane][j] = tgt[(size_t)(b0 + lane) * NQ + j];
    }
    __syncthreads();

    float acc = 0.0f;

    for (int l = 0; l < NLAYER; ++l) {
        // ---- phase 1: softmax stats + raw cost rows ------------------------
        // lane handles rows (b = lane%16, q in {0,1,2} or {3,4,5})
        {
            const int b = lane & 15;
            const int qbase = (lane < 16) ? 0 : 3;
            const float* src =
                outputs + ((size_t)l * BATCH + (size_t)(b0 + b)) * (NQ * NCL);
            #pragma unroll
            for (int qq = 0; qq < 3; ++qq) {
                const int q = qbase + qq;
                const float* row = src + q * NCL;
                float v[NCL];
                #pragma unroll
                for (int j = 0; j < NCL; ++j) v[j] = row[j];
                float mx = v[0];
                #pragma unroll
                for (int j = 1; j < NCL; ++j) mx = fmaxf(mx, v[j]);
                float se = 0.0f, swl = 0.0f;
                #pragma unroll
                for (int j = 0; j < NCL; ++j) {
                    se  += __expf(v[j] - mx);
                    swl += ((j == 9) ? W9 : W0) * v[j];
                }
                const float lse = mx + __logf(se);
                const float inv = 1.0f / se;
                s_lse[wave][b][q] = lse;
                s_swl[wave][b][q] = swl;
                #pragma unroll
                for (int j = 0; j < NCL; ++j) s_logit[wave][b][q * NCL + j] = v[j];
                #pragma unroll
                for (int j = 0; j < NQ; ++j) {
                    const int y = (int)s_tgt[wave][b][j];
                    s_Craw[wave][b][q * NQ + j] = -(__expf(v[y] - mx) * inv);
                }
            }
        }
        __syncthreads();

        // ---- phase 2: A fragment (reduced cost, K=30 pad 32) ---------------
        // A layout (matches B): lanes<16 hold K 0..15,
        // lanes>=16 hold K 16..31; M = lane%16.
        v16h a0;
        {
            const int b = lane & 15;
            const float* cr = &s_Craw[wave][b][0];
            const float* c5 = cr + 30;            // row q=5
            #pragma unroll
            for (int i = 0; i < 16; ++i) a0[i] = (_Float16)0.0f;
            if (lane < 16) {
                #pragma unroll
                for (int i = 0; i < 16; ++i) {
                    const int k = i;                            // 0..15
                    const int q = k / 6, j = k % 6;
                    a0[i] = (_Float16)(cr[q * 6 + j] - c5[j]);
                }
            } else {
                #pragma unroll
                for (int i = 0; i < 16; ++i) {
                    const int k = 16 + i;                       // 16..31
                    if (k < 30) {
                        const int q = k / 6, j = k % 6;
                        a0[i] = (_Float16)(cr[q * 6 + j] - c5[j]);
                    }
                }
            }
        }

        // ---- WMMA permutation scoring + running argmin ---------------------
        float mval[8];
        int   midx[8];
        #pragma unroll
        for (int v = 0; v < 8; ++v) { mval[v] = 3.4e38f; midx[v] = 0; }
        const int nsel = lane & 15;

        for (int t = 0; t < NTILE; ++t) {
            const _Float16* bp = bfrag + (size_t)t * 512 + (size_t)lane * 16;
            __builtin_prefetch(bp + 512, 0, 3);       // next tile's fragment
            const v16h bq = *(const v16h*)bp;
            v8f c = {};
            c = __builtin_amdgcn_wmma_f32_16x16x32_f16(false, a0, false, bq,
                                                       (short)0, c, false, false);
            const int pidx = t * 16 + nsel;
            #pragma unroll
            for (int v = 0; v < 8; ++v) {
                const float cv = c[v];
                if (cv < mval[v]) { mval[v] = cv; midx[v] = pidx; }
            }
        }

        // ---- branchless cross-lane argmin (16-lane groups) -----------------
        // pack (ordered float key, perm index) into u64; min == argmin with
        // smaller-index tie-break (jnp.argmin semantics)
        uint64_t kk[8];
        #pragma unroll
        for (int v = 0; v < 8; ++v)
            kk[v] = ((uint64_t)fkey(mval[v]) << 32) | (uint32_t)midx[v];
        #pragma unroll
        for (int v = 0; v < 8; ++v) {
            uint64_t k = kk[v];
            #pragma unroll
            for (int s = 1; s < 16; s <<= 1) {
                const uint32_t olo = __shfl_xor((int)(uint32_t)k, s, 32);
                const uint32_t ohi = __shfl_xor((int)(uint32_t)(k >> 32), s, 32);
                const uint64_t o = ((uint64_t)ohi << 32) | olo;
                k = (o < k) ? o : k;
            }
            kk[v] = k;
        }

        // ---- phase 3: matched weighted CE with label smoothing -------------
        float sample = 0.0f;
        int v = -1;
        if (lane < 8) v = lane;
        else if (lane >= 16 && lane < 24) v = lane - 16;
        if (v >= 0) {
            const int b = (lane < 16) ? v : v + 8;    // D-tile row mapping
            uint64_t kb = 0;
            #pragma unroll
            for (int u = 0; u < 8; ++u) if (u == v) kb = kk[u];
            const int best = (int)(uint32_t)kb;
            const uint32_t pp = s_perm[best];
            float per = 0.0f, den = 0.0f;
            #pragma unroll
            for (int q = 0; q < NQ; ++q) {
                const int col = (int)((pp >> (4 * q)) & 0xF);
                const int y   = (int)s_tgt[wave][b][col];
                const float wy  = (y == 9) ? W9 : W0;
                const float lse = s_lse[wave][b][q];
                const float ly  = s_logit[wave][b][q * NCL + y];
                const float nll    = -wy * (ly - lse);
                const float smooth = 10.0f * lse - s_swl[wave][b][q]; // sum(w)=10
                per += (1.0f - EPSW) * nll + (EPSW / NCL) * smooth;
                den += wy;
            }
            sample = per / den;
        }
        float ssum = sample;
        #pragma unroll
        for (int s = 1; s < 32; s <<= 1) ssum += __shfl_xor(ssum, s, 32);
        const float coef = (l == NLAYER - 1) ? 1.0f : (0.5f / (NLAYER - 1));
        acc += coef * ssum;

        __syncthreads();   // LDS reuse next layer
    }

    if (lane == 0) s_red[wave] = acc;
    __syncthreads();
    if (tid == 0) {
        const float tot = s_red[0] + s_red[1] + s_red[2] + s_red[3];
        atomicAdd(out, tot * (1.0f / (float)BATCH));
    }
}

// ---------------------------------------------------------------------------
extern "C" void kernel_launch(void* const* d_in, const int* in_sizes, int n_in,
                              void* d_out, int out_size, void* d_ws, size_t ws_size,
                              hipStream_t stream) {
    const float* outputs = (const float*)d_in[0];   // (7,32768,6,10) f32
    const float* targets = (const float*)d_in[1];   // (32768,6,10)   f32
    float* out = (float*)d_out;                     // scalar f32

    char* ws = (char*)d_ws;                         // needs ~432 KB scratch
    _Float16* bf = (_Float16*)(ws + OFF_BF);
    uint32_t* pt = (uint32_t*)(ws + OFF_PERM);
    short*    tg = (short*)(ws + OFF_TGT);

    hml_init<<<90, 256, 0, stream>>>(bf, pt, out);                  // 23040 threads
    hml_tgt<<<(BATCH * NQ + 255) / 256, 256, 0, stream>>>(targets, tg);
    hml_main<<<NBT / 4, 128, 0, stream>>>(outputs, tg, bf, pt, out);
}
